// PointNet2_MC_56710748176470
// MI455X (gfx1250) — matrix operand order repaired and checked by reading, not compiled, not run on hardware
//
#include <hip/hip_runtime.h>
#include <hip/hip_bf16.h>

// ---------------------------------------------------------------------------
// PointNet++ (B=8, N=8192, CIN=6, 24 classes) forward pass for MI455X/gfx1250.
// All dense math runs through V_WMMA_F32_16X16X32_BF16 (bf16 A/B, f32 accum).
// GEMM: one 16(M)x32(N) tile per wave32 -- one A fragment feeds two WMMAs.
// ---------------------------------------------------------------------------

typedef __attribute__((ext_vector_type(16))) __bf16 v16bf;
typedef __attribute__((ext_vector_type(8)))  __bf16 v8bf;
typedef __attribute__((ext_vector_type(8)))  float  v8f;

#define EPSF 1e-5f

static __host__ __device__ inline int rnd32i(int k) { return (k + 31) & ~31; }

// ---------------------------------------------------------------------------
// Input transpose: xin [B, C, N] -> pts [B, N, C] and xyz0 [B, N, 3]
// ---------------------------------------------------------------------------
__global__ void k_transpose_in(const float* __restrict__ xin,
                               float* __restrict__ pts, float* __restrict__ xyz0,
                               int Bt, int C, int N) {
  int i = blockIdx.x * blockDim.x + threadIdx.x;
  if (i >= Bt * C * N) return;
  int n = i % N;
  int c = (i / N) % C;
  int b = i / (N * C);
  float v = xin[i];
  pts[((size_t)b * N + n) * C + c] = v;
  if (c < 3) xyz0[((size_t)b * N + n) * 3 + c] = v;
}

// ---------------------------------------------------------------------------
// Farthest point sampling: one workgroup per batch; running min-distance array
// lives entirely in LDS (N <= 8192 -> 32KB, well inside 320KB/WGP).
// Records `far` *before* updating, matching the reference scan semantics.
// ---------------------------------------------------------------------------
__global__ void k_fps(const float* __restrict__ xyz, int* __restrict__ fidx,
                      int N, int npoint) {
  extern __shared__ float sm[];
  float* dist = sm;                 // N floats
  float* rv   = sm + N;             // blockDim floats
  int*   ri   = (int*)(rv + blockDim.x);
  int b = blockIdx.x, tid = threadIdx.x, nt = blockDim.x;
  const float* px = xyz + (size_t)b * N * 3;
  for (int i = tid; i < N; i += nt) dist[i] = 1e10f;
  __syncthreads();
  int far = 0;
  for (int it = 0; it < npoint; ++it) {
    if (tid == 0) fidx[b * npoint + it] = far;
    float cx = px[far * 3 + 0], cy = px[far * 3 + 1], cz = px[far * 3 + 2];
    float bv = -1.f; int bi = 0;
    for (int i = tid; i < N; i += nt) {
      float dx = px[i * 3 + 0] - cx;
      float dy = px[i * 3 + 1] - cy;
      float dz = px[i * 3 + 2] - cz;
      float d = dx * dx + dy * dy + dz * dz;
      d = fminf(dist[i], d);
      dist[i] = d;
      if (d > bv) { bv = d; bi = i; }
    }
    rv[tid] = bv; ri[tid] = bi;
    __syncthreads();
    for (int s = nt >> 1; s > 0; s >>= 1) {
      if (tid < s) {
        bool take = (rv[tid + s] > rv[tid]) ||
                    (rv[tid + s] == rv[tid] && ri[tid + s] < ri[tid]);
        if (take) { rv[tid] = rv[tid + s]; ri[tid] = ri[tid + s]; }
      }
      __syncthreads();
    }
    far = ri[0];
    __syncthreads();
  }
}

__global__ void k_gather_xyz(const float* __restrict__ xyz, const int* __restrict__ fidx,
                             float* __restrict__ nx, int N, int S, int Bt) {
  int i = blockIdx.x * blockDim.x + threadIdx.x;
  if (i >= Bt * S) return;
  int b = i / S;
  int j = fidx[i];
  const float* p = xyz + ((size_t)b * N + j) * 3;
  nx[(size_t)i * 3 + 0] = p[0];
  nx[(size_t)i * 3 + 1] = p[1];
  nx[(size_t)i * 3 + 2] = p[2];
}

// ---------------------------------------------------------------------------
// Ball query: ascending-index scan == reference's sort + take-first-nsample.
// ---------------------------------------------------------------------------
__global__ void k_ball_query(const float* __restrict__ xyz, const float* __restrict__ nx,
                             int* __restrict__ gidx, int N, int S, float r2,
                             int ns, int Bt) {
  int i = blockIdx.x * blockDim.x + threadIdx.x;
  if (i >= Bt * S) return;
  int b = i / S;
  float cx = nx[(size_t)i * 3 + 0];
  float cy = nx[(size_t)i * 3 + 1];
  float cz = nx[(size_t)i * 3 + 2];
  const float* px = xyz + (size_t)b * N * 3;
  int* out = gidx + (size_t)i * ns;
  int cnt = 0, first = 0;
  bool found = false;
  for (int j = 0; j < N && cnt < ns; ++j) {
    float dx = px[j * 3 + 0] - cx;
    float dy = px[j * 3 + 1] - cy;
    float dz = px[j * 3 + 2] - cz;
    float d = dx * dx + dy * dy + dz * dz;
    if (d <= r2) {
      if (!found) { first = j; found = true; }
      out[cnt++] = j;
    }
  }
  for (; cnt < ns; ++cnt) out[cnt] = first;
}

// ---------------------------------------------------------------------------
// Grouping: build bf16 WMMA A-operand rows [ (b,s,k) , Kp ] =
// [ xyz[idx]-center (3) | points[idx] (C) | zero pad ]
// ---------------------------------------------------------------------------
__global__ void k_group(const float* __restrict__ xyz, const float* __restrict__ nx,
                        const float* __restrict__ pin, const int* __restrict__ gidx,
                        __bf16* __restrict__ A, int N, int S, int ns, int C,
                        int Kp, int Bt) {
  int i = blockIdx.x * blockDim.x + threadIdx.x;
  if (i >= Bt * S * ns) return;
  int sidx = i / ns;          // b*S + s
  int b = sidx / S;
  int j = gidx[i];
  __bf16* dst = A + (size_t)i * Kp;
  const float* pj = xyz + ((size_t)b * N + j) * 3;
  const float* cs = nx + (size_t)sidx * 3;
  dst[0] = (__bf16)(pj[0] - cs[0]);
  dst[1] = (__bf16)(pj[1] - cs[1]);
  dst[2] = (__bf16)(pj[2] - cs[2]);
  const float* fj = pin + ((size_t)b * N + j) * C;
  for (int c = 0; c < C; ++c) dst[3 + c] = (__bf16)fj[c];
  for (int c = 3 + C; c < Kp; ++c) dst[c] = (__bf16)0.f;
}

// ---------------------------------------------------------------------------
// Weight convert: W [O, K] f32 -> Wb [O, Kp] bf16 (zero K-pad). The WMMA
// B-fragment wants contiguous K per output column == a row of W: no transpose.
// ---------------------------------------------------------------------------
__global__ void k_cvt_w(const float* __restrict__ W, __bf16* __restrict__ Wb,
                        int O, int K, int Kp) {
  int i = blockIdx.x * blockDim.x + threadIdx.x;
  if (i >= O * Kp) return;
  int o = i / Kp, k = i % Kp;
  Wb[i] = (__bf16)(k < K ? W[(size_t)o * K + k] : 0.f);
}

// ---------------------------------------------------------------------------
// GEMM + bias via V_WMMA_F32_16X16X32_BF16. One 16(M)x32(N) tile per wave32:
// a single A fragment is reused by two WMMAs (two B fragments / accumulators),
// halving A traffic per matrix op. 8 waves/block = 4 M-tiles x 2 N-pairs =
// 64(M) x 64(N) per block. Fragment element order is contiguous k = 16h + s:
// A and B lanes 0-15/16-31 both split K at +16; D lane=N, vgpr=M (+8 for
// upper lanes). EXEC is all-ones through the WMMA loop (wave-uniform exit
// only; per-lane column guards affect pointer selection and stores only).
// ---------------------------------------------------------------------------
__global__ void k_gemm_wmma(const __bf16* __restrict__ A, const __bf16* __restrict__ W,
                            const float* __restrict__ bias, float* __restrict__ Z,
                            int M, int Kp, int O) {
  int lane = threadIdx.x & 31;
  int wave = threadIdx.x >> 5;
  int m0 = (blockIdx.x * 4 + (wave & 3)) * 16;
  int n0 = (blockIdx.y * 2 + (wave >> 2)) * 32;
  if (m0 >= M) return;                       // wave-uniform exit
  int half = lane >> 4;                      // 0: lanes 0-15, 1: lanes 16-31
  int l15  = lane & 15;
  const __bf16* arow = A + (size_t)(m0 + l15) * Kp + (half << 4);   // +0 / +16 K
  int col0 = n0 + l15;
  int col1 = n0 + 16 + l15;
  const __bf16* wrow0 = W + (size_t)(col0 < O ? col0 : 0) * Kp + (half << 4);
  const __bf16* wrow1 = W + (size_t)(col1 < O ? col1 : 0) * Kp + (half << 4);
  v8f acc0 = {}, acc1 = {};
  for (int k = 0; k < Kp; k += 32) {
    __builtin_prefetch(arow + k + 64, 0, 1);   // global_prefetch next K tiles
    v8bf a0  = *(const v8bf*)(arow + k);
    v8bf a1  = *(const v8bf*)(arow + k + 8);
    v8bf b00 = *(const v8bf*)(wrow0 + k);
    v8bf b01 = *(const v8bf*)(wrow0 + k + 8);
    v8bf b10 = *(const v8bf*)(wrow1 + k);
    v8bf b11 = *(const v8bf*)(wrow1 + k + 8);
    v16bf av, bv0, bv1;
#pragma unroll
    for (int t = 0; t < 8; ++t) {
      av[t]  = a0[t];  av[t + 8]  = a1[t];
      bv0[t] = b00[t]; bv0[t + 8] = b01[t];
      bv1[t] = b10[t]; bv1[t + 8] = b11[t];
    }
    acc0 = __builtin_amdgcn_wmma_f32_16x16x32_bf16(
        false, av, false, bv0, (short)0, acc0, false, false);
    acc1 = __builtin_amdgcn_wmma_f32_16x16x32_bf16(
        false, av, false, bv1, (short)0, acc1, false, false);
  }
  int mr = m0 + half * 8;
  if (col0 < O) {
    float bb = bias ? bias[col0] : 0.f;
#pragma unroll
    for (int t = 0; t < 8; ++t)
      Z[(size_t)(mr + t) * O + col0] = acc0[t] + bb;
  }
  if (col1 < O) {
    float bb = bias ? bias[col1] : 0.f;
#pragma unroll
    for (int t = 0; t < 8; ++t)
      Z[(size_t)(mr + t) * O + col1] = acc1[t] + bb;
  }
}

// ---------------------------------------------------------------------------
// BatchNorm statistics over all M points (axes (0,1,2) of reference), one
// block per channel. Z (<= 67MB) stays resident in the 192MB L2, so the
// column-strided re-reads across channel-blocks are L2 hits.
// ---------------------------------------------------------------------------
__global__ void k_bn_stats(const float* __restrict__ Z, float* __restrict__ scale,
                           float* __restrict__ shift, const float* __restrict__ g,
                           const float* __restrict__ beta, int M, int O) {
  __shared__ float s1[256], s2[256];
  int o = blockIdx.x, tid = threadIdx.x;
  float a = 0.f, b = 0.f;
  for (int m = tid; m < M; m += blockDim.x) {
    float v = Z[(size_t)m * O + o];
    a += v; b += v * v;
  }
  s1[tid] = a; s2[tid] = b;
  __syncthreads();
  for (int s = 128; s > 0; s >>= 1) {
    if (tid < s) { s1[tid] += s1[tid + s]; s2[tid] += s2[tid + s]; }
    __syncthreads();
  }
  if (tid == 0) {
    float mean = s1[0] / (float)M;
    float var  = s2[0] / (float)M - mean * mean;
    float sc = g[o] * rsqrtf(var + EPSF);
    scale[o] = sc;
    shift[o] = beta[o] - mean * sc;
  }
}

// ---------------------------------------------------------------------------
// Fused BN+ReLU; optionally writes f32 (may alias Z) and the zero-padded bf16
// A-operand for the next GEMM.
// ---------------------------------------------------------------------------
__global__ void k_bn_relu(const float* __restrict__ Z, float* __restrict__ outF,
                          __bf16* __restrict__ Anext, const float* __restrict__ scale,
                          const float* __restrict__ shift, int M, int O, int Kpn) {
  int so = Anext ? Kpn : O;
  int i = blockIdx.x * blockDim.x + threadIdx.x;
  if (i >= M * so) return;
  int o = i % so, m = i / so;
  if (o < O) {
    float v = Z[(size_t)m * O + o] * scale[o] + shift[o];
    v = fmaxf(v, 0.f);
    if (outF)  outF[(size_t)m * O + o] = v;
    if (Anext) Anext[i] = (__bf16)v;
  } else if (Anext) {
    Anext[i] = (__bf16)0.f;
  }
}

// Max-pool over the nsample axis: Y [MS, ns, O] -> out [MS, O]
__global__ void k_max_pool(const float* __restrict__ Y, float* __restrict__ out,
                           int MS, int ns, int O) {
  int i = blockIdx.x * blockDim.x + threadIdx.x;
  if (i >= MS * O) return;
  int o = i % O, s = i / O;
  const float* p = Y + ((size_t)s * ns) * O + o;
  float m = p[0];
  for (int k = 1; k < ns; ++k) m = fmaxf(m, p[(size_t)k * O]);
  out[i] = m;
}

// ---------------------------------------------------------------------------
// FP stage: 3-NN inverse-distance interpolation + concat, emitting the bf16
// A-operand [B*n1, Kp] = [ points1 (C1) | interp (C2) | pad ].
// ---------------------------------------------------------------------------
__global__ void k_fp_interp(const float* __restrict__ x1, const float* __restrict__ x2,
                            const float* __restrict__ p1, int C1,
                            const float* __restrict__ p2, int C2,
                            __bf16* __restrict__ A, int Kp, int n1, int n2, int Bt) {
  int i = blockIdx.x * blockDim.x + threadIdx.x;
  if (i >= Bt * n1) return;
  int b = i / n1;
  const float* q = x1 + (size_t)i * 3;
  float qx = q[0], qy = q[1], qz = q[2];
  float d0 = 3e38f, d1 = 3e38f, d2 = 3e38f;
  int j0 = 0, j1 = 0, j2 = 0;
  const float* c = x2 + (size_t)b * n2 * 3;
  for (int j = 0; j < n2; ++j) {
    float dx = c[j * 3 + 0] - qx;
    float dy = c[j * 3 + 1] - qy;
    float dz = c[j * 3 + 2] - qz;
    float d = dx * dx + dy * dy + dz * dz;
    if (d < d0)      { d2 = d1; j2 = j1; d1 = d0; j1 = j0; d0 = d; j0 = j; }
    else if (d < d1) { d2 = d1; j2 = j1; d1 = d;  j1 = j; }
    else if (d < d2) { d2 = d;  j2 = j; }
  }
  float w0 = 1.f / (d0 + 1e-8f), w1 = 1.f / (d1 + 1e-8f), w2 = 1.f / (d2 + 1e-8f);
  float wsum = w0 + w1 + w2;
  w0 /= wsum; w1 /= wsum; w2 /= wsum;
  __bf16* dst = A + (size_t)i * Kp;
  for (int cc = 0; cc < C1; ++cc) dst[cc] = (__bf16)p1[(size_t)i * C1 + cc];
  const float* r0 = p2 + ((size_t)b * n2 + j0) * C2;
  const float* r1 = p2 + ((size_t)b * n2 + j1) * C2;
  const float* r2 = p2 + ((size_t)b * n2 + j2) * C2;
  for (int cc = 0; cc < C2; ++cc)
    dst[C1 + cc] = (__bf16)(w0 * r0[cc] + w1 * r1[cc] + w2 * r2[cc]);
  for (int cc = C1 + C2; cc < Kp; ++cc) dst[cc] = (__bf16)0.f;
}

// Final transpose: Z [B*N, NC] -> out [B, NC, N]
__global__ void k_write_logits(const float* __restrict__ Z, float* __restrict__ out,
                               int Bt, int N, int NC) {
  int i = blockIdx.x * blockDim.x + threadIdx.x;
  if (i >= Bt * N * NC) return;
  int o = i % NC;
  int n = (i / NC) % N;
  int b = i / (NC * N);
  out[((size_t)b * NC + o) * N + n] = Z[i];
}

// ---------------------------------------------------------------------------
// Host orchestration
// ---------------------------------------------------------------------------
extern "C" void kernel_launch(void* const* d_in, const int* in_sizes, int n_in,
                              void* d_out, int out_size, void* d_ws, size_t ws_size,
                              hipStream_t stream) {
  (void)in_sizes; (void)n_in; (void)out_size; (void)ws_size;
  const int Bc = 8, N = 8192, CIN = 6, NC = 24;

  // ---- input pointer walk (dict order: xyz, sa1..sa4, fp4..fp1, head, cls)
  int idx = 0;
  const float* xin = (const float*)d_in[idx++];
  struct L { const float *W, *b, *g, *beta; int K, O; };
  auto grab = [&](int K, int O) -> L {
    L l;
    l.W = (const float*)d_in[idx++]; l.b = (const float*)d_in[idx++];
    l.g = (const float*)d_in[idx++]; l.beta = (const float*)d_in[idx++];
    l.K = K; l.O = O; return l;
  };
  L sa1[3] = { grab(9, 32),   grab(32, 32),  grab(32, 64)  };
  L sa2[3] = { grab(67, 64),  grab(64, 64),  grab(64, 128) };
  L sa3[3] = { grab(131, 128),grab(128, 128),grab(128, 256)};
  L sa4[3] = { grab(259, 256),grab(256, 256),grab(256, 512)};
  L fp4[2] = { grab(768, 256),grab(256, 256) };
  L fp3[2] = { grab(384, 256),grab(256, 256) };
  L fp2[2] = { grab(320, 256),grab(256, 128) };
  L fp1[3] = { grab(128, 128),grab(128, 128),grab(128, 128) };
  L headl  = grab(128, 128);
  const float* clsW = (const float*)d_in[idx++];
  const float* clsb = (const float*)d_in[idx++];

  // ---- workspace bump allocator (~115 MB total)
  char* wp = (char*)d_ws;
  auto alloc = [&](size_t bytes) -> char* {
    char* p = wp; wp += (bytes + 255) & ~(size_t)255; return p;
  };
  float* pts  = (float*)alloc((size_t)Bc * N * CIN * 4);
  float* xyz0 = (float*)alloc((size_t)Bc * N * 3 * 4);
  float* l1x  = (float*)alloc((size_t)Bc * 1024 * 3 * 4);
  float* l2x  = (float*)alloc((size_t)Bc * 256 * 3 * 4);
  float* l3x  = (float*)alloc((size_t)Bc * 64 * 3 * 4);
  float* l4x  = (float*)alloc((size_t)Bc * 16 * 3 * 4);
  float* l1p  = (float*)alloc((size_t)Bc * 1024 * 64 * 4);
  float* l2p  = (float*)alloc((size_t)Bc * 256 * 128 * 4);
  float* l3p  = (float*)alloc((size_t)Bc * 64 * 256 * 4);
  float* l4p  = (float*)alloc((size_t)Bc * 16 * 512 * 4);
  float* l3pn = (float*)alloc((size_t)Bc * 64 * 256 * 4);
  float* l2pn = (float*)alloc((size_t)Bc * 256 * 256 * 4);
  float* l1pn = (float*)alloc((size_t)Bc * 1024 * 128 * 4);
  int*   fidx = (int*)alloc((size_t)Bc * 1024 * 4);
  int*   gidx = (int*)alloc((size_t)Bc * 1024 * 32 * 4);
  float* Sc   = (float*)alloc(512 * 4);
  float* Sh   = (float*)alloc(512 * 4);
  __bf16* Wb  = (__bf16*)alloc((size_t)262144 * 2);
  __bf16* A0  = (__bf16*)alloc((size_t)8388608 * 2);
  __bf16* A1  = (__bf16*)alloc((size_t)8388608 * 2);
  float* Zb   = (float*)alloc((size_t)16777216 * 4);

  // ---- one linear layer: GEMM(bf16)+bias -> BN stats -> BN+ReLU (+bf16 repack)
  auto run_layer = [&](const __bf16* Ain, int M, const L& l,
                       float* outF, __bf16* Anext) {
    int Kp = rnd32i(l.K), Kpn = rnd32i(l.O);
    int wn = l.O * Kp;
    k_cvt_w<<<(wn + 255) / 256, 256, 0, stream>>>(l.W, Wb, l.O, l.K, Kp);
    dim3 gg((M + 63) / 64, (l.O + 63) / 64);
    k_gemm_wmma<<<gg, 256, 0, stream>>>(Ain, Wb, l.b, Zb, M, Kp, l.O);
    k_bn_stats<<<l.O, 256, 0, stream>>>(Zb, Sc, Sh, l.g, l.beta, M, l.O);
    int tot = Anext ? M * Kpn : M * l.O;
    k_bn_relu<<<(tot + 255) / 256, 256, 0, stream>>>(Zb, outF, Anext, Sc, Sh, M, l.O, Kpn);
  };

  // ---- set-abstraction stage
  auto run_sa = [&](const float* xyzin, int Nin, const float* pin, int C,
                    int S, float radius, int ns, const L* ls, int nl,
                    float* xyzout, float* pooled) {
    size_t shm = (size_t)(Nin + 256) * 4 + 256 * 4;
    k_fps<<<Bc, 256, shm, stream>>>(xyzin, fidx, Nin, S);
    int q = Bc * S;
    k_gather_xyz<<<(q + 255) / 256, 256, 0, stream>>>(xyzin, fidx, xyzout, Nin, S, Bc);
    k_ball_query<<<(q + 255) / 256, 256, 0, stream>>>(xyzin, xyzout, gidx, Nin, S,
                                                      radius * radius, ns, Bc);
    int M = Bc * S * ns;
    int Kp0 = rnd32i(3 + C);
    k_group<<<(M + 255) / 256, 256, 0, stream>>>(xyzin, xyzout, pin, gidx, A0,
                                                 Nin, S, ns, C, Kp0, Bc);
    const __bf16* cur = A0;
    for (int i = 0; i < nl; ++i) {
      bool last = (i == nl - 1);
      __bf16* nxt = (cur == A0) ? A1 : A0;
      run_layer(cur, M, ls[i], Zb /* f32 in-place */, last ? nullptr : nxt);
      if (!last) cur = nxt;
    }
    int Ol = ls[nl - 1].O;
    k_max_pool<<<(Bc * S * Ol + 255) / 256, 256, 0, stream>>>(Zb, pooled, Bc * S, ns, Ol);
  };

  // ---- feature-propagation stage; returns bf16 buffer holding final features
  auto run_fp = [&](const float* x1, int n1, const float* x2, int n2,
                    const float* p1, int C1, const float* p2, int C2,
                    const L* ls, int nl, float* outF) -> const __bf16* {
    int M = Bc * n1;
    int Kp0 = rnd32i(C1 + C2);
    k_fp_interp<<<(M + 255) / 256, 256, 0, stream>>>(x1, x2, p1, C1, p2, C2,
                                                     A0, Kp0, n1, n2, Bc);
    const __bf16* cur = A0;
    for (int i = 0; i < nl; ++i) {
      bool last = (i == nl - 1);
      __bf16* nxt = (cur == A0) ? A1 : A0;
      run_layer(cur, M, ls[i], last ? outF : Zb, nxt);
      cur = nxt;
    }
    return cur;
  };

  // ---- pipeline ----
  {
    int tot = Bc * CIN * N;
    k_transpose_in<<<(tot + 255) / 256, 256, 0, stream>>>(xin, pts, xyz0, Bc, CIN, N);
  }
  run_sa(xyz0, N,    pts, CIN, 1024, 0.1f, 32, sa1, 3, l1x, l1p);
  run_sa(l1x,  1024, l1p, 64,  256,  0.2f, 32, sa2, 3, l2x, l2p);
  run_sa(l2x,  256,  l2p, 128, 64,   0.4f, 32, sa3, 3, l3x, l3p);
  run_sa(l3x,  64,   l3p, 256, 16,   0.8f, 32, sa4, 3, l4x, l4p);

  run_fp(l3x, 64,   l4x, 16,   l3p, 256, l4p,  512, fp4, 2, l3pn);
  run_fp(l2x, 256,  l3x, 64,   l2p, 128, l3pn, 256, fp3, 2, l2pn);
  run_fp(l1x, 1024, l2x, 256,  l1p, 64,  l2pn, 256, fp2, 2, l1pn);
  const __bf16* l0a =
      run_fp(xyz0, N, l1x, 1024, nullptr, 0, l1pn, 128, fp1, 3, Zb);

  // head (BN+ReLU) -> bf16
  __bf16* hx = (l0a == A0) ? A1 : A0;
  run_layer(l0a, Bc * N, headl, Zb, hx);

  // classifier: plain GEMM + bias, then transpose to (B, NC, N)
  {
    int M = Bc * N, Kp = rnd32i(128);
    int wn = NC * Kp;
    k_cvt_w<<<(wn + 255) / 256, 256, 0, stream>>>(clsW, Wb, NC, 128, Kp);
    dim3 gg((M + 63) / 64, (NC + 63) / 64);
    k_gemm_wmma<<<gg, 256, 0, stream>>>(hx, Wb, clsb, Zb, M, Kp, NC);
    k_write_logits<<<(M * NC + 255) / 256, 256, 0, stream>>>(Zb, (float*)d_out, Bc, N, NC);
  }
}
